// _IntEinsum_57286273794907
// MI455X (gfx1250) — matrix operand order repaired and checked by reading, not compiled, not run on hardware
//
#include <hip/hip_runtime.h>
#include <cstdint>

// ---- CDNA5 (gfx1250) wave32 types ----
typedef __attribute__((ext_vector_type(16))) __bf16 v16bf;
typedef __attribute__((ext_vector_type(4)))  __bf16 v4bf;
typedef __attribute__((ext_vector_type(8)))  float  v8f;
typedef __attribute__((ext_vector_type(4)))  float  v4f;
typedef __attribute__((ext_vector_type(4)))  unsigned int u32x4;
typedef __attribute__((ext_vector_type(8)))  int    i32x8;
typedef __attribute__((ext_vector_type(4)))  int    i32x4;

#define AS1 __attribute__((address_space(1)))
#define AS3 __attribute__((address_space(3)))

#if __has_builtin(__builtin_amdgcn_tensor_load_to_lds)
  #define USE_TDM 1
#elif __has_builtin(__builtin_amdgcn_global_load_async_to_lds_b128)
  #define USE_ASYNC 1
#endif

constexpr int M_TOT = 8192;   // B*T
constexpr int K_TOT = 4096;   // D
constexpr int N_TOT = 16384;  // F
constexpr int BM = 128, BN = 128, BK = 64;

__device__ __forceinline__ unsigned lds_byte_off(const void* p) {
  return (unsigned)(size_t)(AS3 const char*)p;  // LDS offset via addrspace(3) ptrtoint
}

__device__ __forceinline__ void stage_wait() {
#if defined(USE_TDM)
  __builtin_amdgcn_s_wait_tensorcnt(0);
#elif defined(USE_ASYNC)
 #if __has_builtin(__builtin_amdgcn_s_wait_asynccnt)
  __builtin_amdgcn_s_wait_asynccnt(0);
 #else
  asm volatile("s_wait_asynccnt 0x0" ::: "memory");
 #endif
#endif
}

// Exact truncation split: x = hi + lo, hi has pure bit-extract cost, |err| ~ 2^-16 |x|.
__device__ __forceinline__ void split_bf16(float f, __bf16& h, __bf16& l) {
  const unsigned u = __builtin_bit_cast(unsigned, f);
  h = __builtin_bit_cast(__bf16, (unsigned short)(u >> 16));
  const float hf = __builtin_bit_cast(float, u & 0xFFFF0000u);
  l = (__bf16)(f - hf);
}

#if defined(USE_TDM)
// One TDM descriptor copies the whole 128x64 f32 A tile (row stride = K) to LDS.
__device__ __forceinline__ void tdm_load_A(const float* gtile, unsigned lds_addr) {
  unsigned long long ga = (unsigned long long)(size_t)gtile;
  u32x4 g0;
  g0[0] = 1u;                                                  // count=1 (valid user D#)
  g0[1] = lds_addr;                                            // lds_addr (bytes)
  g0[2] = (unsigned)ga;                                        // global_addr[31:0]
  g0[3] = (unsigned)((ga >> 32) & 0x01FFFFFFull) | (2u << 30); // global_addr[56:32] | type=2
  i32x8 g1;
  g1[0] = (int)(2u << 16);                                     // data_size=2 -> 4-byte elems
  g1[1] = (int)((K_TOT & 0xFFFF) << 16);                       // tensor_dim0[15:0]
  g1[2] = (int)(((K_TOT >> 16) & 0xFFFF) | ((M_TOT & 0xFFFF) << 16)); // dim0 hi | dim1 lo
  g1[3] = (int)(((M_TOT >> 16) & 0xFFFF) | (BK << 16));        // dim1 hi | tile_dim0=BK
  g1[4] = BM;                                                  // tile_dim1=BM, tile_dim2=0
  g1[5] = K_TOT;                                               // tensor_dim0_stride[31:0]
  g1[6] = 0;                                                   // stride0 hi | stride1 lo
  g1[7] = 0;                                                   // stride1 hi
  i32x4 z4 = {0, 0, 0, 0};
#if __clang_major__ >= 23
  i32x8 z8 = {0, 0, 0, 0, 0, 0, 0, 0};
  __builtin_amdgcn_tensor_load_to_lds(g0, g1, z4, z4, z8, 0);
#else
  __builtin_amdgcn_tensor_load_to_lds(g0, g1, z4, z4, 0);
#endif
}
#endif

// Stage the A tile (raw f32) for K-slice k0 into dstA (LDS [BM][BK]).
__device__ __forceinline__ void stage_A(const float* __restrict__ X, int mblk, int k0,
                                        float* dstA, int tid) {
#if defined(USE_TDM)
  if (tid == 0) tdm_load_A(X + (size_t)mblk * K_TOT + k0, lds_byte_off(dstA));
#elif defined(USE_ASYNC)
#pragma unroll
  for (int i = 0; i < 8; ++i) {
    const int chunk = tid + i * 256;              // 16B chunks, 16 per 64-f32 row
    const int row = chunk >> 4, kc = (chunk & 15) * 4;
    const float* g = X + (size_t)(mblk + row) * K_TOT + k0 + kc;
    __builtin_amdgcn_global_load_async_to_lds_b128(
        (AS1 void*)g, (AS3 void*)(dstA + row * BK + kc), 0, 0);
  }
#else
  const int row = tid >> 1, c = (tid & 1) * 32;
  const float* g = X + (size_t)(mblk + row) * K_TOT + k0 + c;
  float* l = dstA + row * BK + c;
#pragma unroll
  for (int i = 0; i < 8; ++i) *(v4f*)(l + i * 4) = *(const v4f*)(g + i * 4);
#endif
}

// B staging: thread covers 4 K rows x 8 N cols of the 64x128 int8 tile.
__device__ __forceinline__ void stage_B_load(const int8_t* __restrict__ W, int nblk, int k0,
                                             int tid, int2* braw) {
  const int bk4 = (tid >> 4) * 4;
  const int bn8 = (tid & 15) * 8;
  const int8_t* g = W + (size_t)(k0 + bk4) * N_TOT + nblk + bn8;
#pragma unroll
  for (int kk = 0; kk < 4; ++kk) braw[kk] = *(const int2*)(g + (size_t)kk * N_TOT);
}

__device__ __forceinline__ void stage_B_store(__bf16* dstB, int tid, const int2* braw) {
  const int bk4 = (tid >> 4) * 4;
  const int bn8 = (tid & 15) * 8;
  const int8_t* p = (const int8_t*)braw;          // p[kk*8 + j]
#pragma unroll
  for (int j = 0; j < 8; ++j) {
    v4bf c;
#pragma unroll
    for (int kk = 0; kk < 4; ++kk) c[kk] = (__bf16)(float)p[kk * 8 + j];  // int8 exact in bf16
    *(v4bf*)&dstB[(bn8 + j) * BK + bk4] = c;      // [N][K] layout, K contiguous, 8B store
  }
}

__global__ __launch_bounds__(256)
void int8w_f32a_gemm_wmma(const float* __restrict__ X,
                          const int8_t* __restrict__ W,
                          const float* __restrict__ S,
                          float* __restrict__ O) {
  __shared__ __align__(32) float  sAf[2][BM * BK];   // raw f32 A, TDM destination (2x32KB)
  __shared__ __align__(32) __bf16 sB [2][BN * BK];   // bf16 B, [N][K]          (2x16KB)

  const int tid  = threadIdx.x;
  const int lane = tid & 31;
  const int wave = tid >> 5;
  const int wm = wave & 3;     // 4 x 32 rows
  const int wn = wave >> 2;    // 2 x 64 cols
  const int nblk = blockIdx.x * BN;
  const int mblk = blockIdx.y * BM;

  const v8f vz = {0.f, 0.f, 0.f, 0.f, 0.f, 0.f, 0.f, 0.f};
  v8f acc[2][4];
#pragma unroll
  for (int mi = 0; mi < 2; ++mi)
#pragma unroll
    for (int ni = 0; ni < 4; ++ni) acc[mi][ni] = vz;

  // ---- prologue: fill buffer 0 ----
  stage_A(X, mblk, 0, &sAf[0][0], tid);
  {
    int2 braw[4];
    stage_B_load(W, nblk, 0, tid, braw);
    stage_B_store(&sB[0][0], tid, braw);
  }
  stage_wait();
  __syncthreads();

  int cur = 0;
  for (int k0 = 0; k0 < K_TOT; k0 += BK) {
    const int nxt = cur ^ 1;
    const bool more = (k0 + BK) < K_TOT;

    int2 braw[4];
    if (more) {
      stage_A(X, mblk, k0 + BK, &sAf[nxt][0], tid);   // TDM DMA overlaps compute
      stage_B_load(W, nblk, k0 + BK, tid, braw);
      if (k0 + 2 * BK < K_TOT)
        __builtin_prefetch(W + (size_t)(k0 + 2 * BK) * N_TOT + nblk + (tid & 15) * 8, 0, 2);
    }

    // ---- compute from buffer `cur`: 2 k-steps of 32, 16 WMMA each ----
    const float*  curA = &sAf[cur][0];
    const __bf16* curB = &sB[cur][0];
    const int mr = lane & 15;

#pragma unroll
    for (int kk = 0; kk < BK; kk += 32) {
      const int akb = kk + ((lane >> 4) << 4);   // A K-offset: +0 / +16 by lane half
      const int bkb = kk + ((lane >> 4) << 4);   // B K-offset: +0 / +16 by lane half

      v16bf ah[2], al[2], bfr[4];
#pragma unroll
      for (int mi = 0; mi < 2; ++mi) {
        const float* pa = curA + (wm * 32 + mi * 16 + mr) * BK + akb;
        float f[16];
        *(v4f*)&f[0]  = *(const v4f*)(pa);
        *(v4f*)&f[4]  = *(const v4f*)(pa + 4);
        *(v4f*)&f[8]  = *(const v4f*)(pa + 8);
        *(v4f*)&f[12] = *(const v4f*)(pa + 12);
#pragma unroll
        for (int j = 0; j < 16; ++j) {           // bit-exact truncation split
          __bf16 h, l;
          split_bf16(f[j], h, l);
          ah[mi][j] = h;
          al[mi][j] = l;
        }
      }
#pragma unroll
      for (int ni = 0; ni < 4; ++ni)
        bfr[ni] = *(const v16bf*)(curB + (wn * 64 + ni * 16 + mr) * BK + bkb);

#pragma unroll
      for (int mi = 0; mi < 2; ++mi)
#pragma unroll
        for (int ni = 0; ni < 4; ++ni) {
          acc[mi][ni] = __builtin_amdgcn_wmma_f32_16x16x32_bf16(
              false, ah[mi], false, bfr[ni], (short)0, acc[mi][ni], false, false);
          acc[mi][ni] = __builtin_amdgcn_wmma_f32_16x16x32_bf16(
              false, al[mi], false, bfr[ni], (short)0, acc[mi][ni], false, false);
        }
    }

    if (more) stage_B_store(&sB[nxt][0], tid, braw);
    stage_wait();
    __syncthreads();
    cur = nxt;
  }

  // ---- epilogue: apply 1/scale, scatter per C layout ----
  const int col0 = nblk + wn * 64 + (lane & 15);
  const int rowh = mblk + wm * 32 + ((lane >> 4) << 3);
#pragma unroll
  for (int ni = 0; ni < 4; ++ni) {
    const int col = col0 + ni * 16;
    const float rs = 1.0f / S[col];
#pragma unroll
    for (int mi = 0; mi < 2; ++mi) {
      const int row = rowh + mi * 16;
#pragma unroll
      for (int r = 0; r < 8; ++r)
        O[(size_t)(row + r) * N_TOT + col] = acc[mi][ni][r] * rs;
    }
  }
}

extern "C" void kernel_launch(void* const* d_in, const int* in_sizes, int n_in,
                              void* d_out, int out_size, void* d_ws, size_t ws_size,
                              hipStream_t stream) {
  (void)in_sizes; (void)n_in; (void)out_size; (void)d_ws; (void)ws_size;
  const float*  x      = (const float*)d_in[0];
  const int8_t* kernel = (const int8_t*)d_in[1];
  const float*  scale  = (const float*)d_in[2];
  float*        out    = (float*)d_out;

  dim3 grid(N_TOT / BN, M_TOT / BM);  // 128 x 64
  dim3 block(256);
  int8w_f32a_gemm_wmma<<<grid, block, 0, stream>>>(x, kernel, scale, out);
}
